// LSTM_55490977464644
// MI455X (gfx1250) — compile-verified, matrix operand order checked by reading, not run on hardware
//
#include <hip/hip_runtime.h>
#include <hip/hip_bf16.h>
#include <math.h>

typedef __attribute__((ext_vector_type(16))) __bf16 v16bf;
typedef __attribute__((ext_vector_type(8)))  __bf16 v8bf;
typedef __attribute__((ext_vector_type(8)))  float  v8f;
typedef __attribute__((ext_vector_type(4)))  float  v4f;

#define B_   128
#define S_   512
#define I_   128
#define H_   256
#define G4_  1024   // 4*H
#define O_   128
#define HPAD 264    // padded h row stride (bf16 elems) -> 528B, 16B aligned, bank-friendly

// Build a 16x32 bf16 WMMA A/B fragment from two contiguous 8-element (16B) pieces,
// per the CDNA5 ISA VGPR layout (lanes 0-15: K {0..7,16..23}; lanes 16-31: K {8..15,24..31}).
__device__ __forceinline__ v16bf load_frag(const __bf16* p0, const __bf16* p1) {
  v8bf lo = *(const v8bf*)p0;
  v8bf hi = *(const v8bf*)p1;
  v16bf r;
#pragma unroll
  for (int i = 0; i < 8; ++i) { r[i] = lo[i]; r[i + 8] = hi[i]; }
  return r;
}

__device__ __forceinline__ float sigf(float x) { return 1.f / (1.f + __expf(-x)); }

// ---------------- fp32 -> bf16 conversion ----------------
__global__ void k_cvt_bf16(const float* __restrict__ in, __bf16* __restrict__ out, int n) {
  int i = blockIdx.x * blockDim.x + threadIdx.x;
  if (i < n) out[i] = (__bf16)in[i];
}

// ---------------- xg = x @ W_ih^T + (b_ih + b_hh) ----------------
// One wave per 16x16 tile; M = B*S = 65536, N = 4H = 1024, K = I = 128.
__global__ void k_xg_gemm(const __bf16* __restrict__ xbf,   // [B*S, I]
                          const __bf16* __restrict__ wih,   // [4H, I]
                          const float*  __restrict__ bih,
                          const float*  __restrict__ bhh,
                          float*        __restrict__ xg)    // [B*S, 4H]
{
  const int lane = threadIdx.x & 31;
  const int wv   = threadIdx.x >> 5;
  const int gid  = blockIdx.x * 8 + wv;
  const int M0   = (gid >> 6) * 16;   // 4096 M tiles
  const int N0   = (gid & 63) * 16;   // 64  N tiles
  const int half = lane >> 4;
  const int l15  = lane & 15;
  const int kaoff = half * 8;
  const int kboff = kaoff + 16;

  const __bf16* arow = xbf + (size_t)(M0 + l15) * I_;
  const __bf16* brow = wih + (size_t)(N0 + l15) * I_;

  v8f acc = {};
#pragma unroll
  for (int kc = 0; kc < 4; ++kc) {
    const int k0 = kc * 32;
    v16bf a = load_frag(arow + k0 + kaoff, arow + k0 + kboff);
    v16bf b = load_frag(brow + k0 + kaoff, brow + k0 + kboff);
    acc = __builtin_amdgcn_wmma_f32_16x16x32_bf16(false, a, false, b, (short)0, acc, false, false);
  }
  const int n = N0 + l15;
  const float bias = bih[n] + bhh[n];
#pragma unroll
  for (int r = 0; r < 8; ++r) {
    const int m = M0 + r + half * 8;
    xg[(size_t)m * G4_ + n] = acc[r] + bias;
  }
}

// ---------------- sequential LSTM scan ----------------
// 8 workgroups (one per 16-row batch tile), 256 threads = 8 waves.
// Each wave computes 128 gate columns via WMMA over K=256; W_hh streams from L2.
// xg loads for step t are issued at the top of the step (b128 vector loads) so
// their global latency hides under the WMMA phase; xg for t+1 is prefetched.
__global__ void __launch_bounds__(256, 1)
k_lstm_scan(const float*  __restrict__ xg,    // [B*S, 4H] (biases included)
            const __bf16* __restrict__ whh,   // [4H, H]
            __bf16*       __restrict__ hs)    // [B, S, H]
{
  __shared__ __bf16 h_lds[16 * HPAD];   // current hidden state (A matrix source)
  __shared__ __bf16 g_lds[16 * G4_];    // pre-activation gates (h@Whh^T part)

  const int tid  = threadIdx.x;
  const int lane = tid & 31;
  const int wv   = tid >> 5;
  const int half = lane >> 4;
  const int l15  = lane & 15;
  const int kaoff = half * 8;
  const int kboff = kaoff + 16;
  const int b0 = blockIdx.x * 16;       // batch rows [b0, b0+16)

  // fixed elementwise mapping: thread -> (row er, 16 consecutive cols at ec)
  const int er = tid >> 4;
  const int ec = (tid & 15) * 16;
  float c_st[16];
#pragma unroll
  for (int j = 0; j < 16; ++j) c_st[j] = 0.f;

  for (int i = tid; i < 16 * HPAD; i += 256) h_lds[i] = (__bf16)0.f;
  __syncthreads();

  for (int t = 0; t < S_; ++t) {
    // ---- issue this step's xg loads early (overlap with WMMA phase) ----
    const size_t xgrow = ((size_t)(b0 + er) * S_ + t) * G4_;
    v4f xv[4][4];   // [gate][4 x float4] = 16 floats per gate
#pragma unroll
    for (int g = 0; g < 4; ++g) {
      const v4f* p = (const v4f*)(xg + xgrow + g * 256 + ec);
#pragma unroll
      for (int q = 0; q < 4; ++q) xv[g][q] = p[q];
    }

    // ---- recurrent GEMM: gates_partial = h @ Whh^T ----
    v16bf af[8];
#pragma unroll
    for (int kc = 0; kc < 8; ++kc) {
      const __bf16* base = &h_lds[l15 * HPAD + kc * 32];
      af[kc] = load_frag(base + kaoff, base + kboff);
    }
#pragma unroll
    for (int nt = 0; nt < 8; ++nt) {
      const int n0 = wv * 128 + nt * 16;
      const __bf16* brow = whh + (size_t)(n0 + l15) * H_;
      v8f acc = {};
#pragma unroll
      for (int kc = 0; kc < 8; ++kc) {
        v16bf b = load_frag(brow + kc * 32 + kaoff, brow + kc * 32 + kboff);
        acc = __builtin_amdgcn_wmma_f32_16x16x32_bf16(false, af[kc], false, b, (short)0, acc, false, false);
      }
      const int n = n0 + l15;
#pragma unroll
      for (int r = 0; r < 8; ++r)
        g_lds[(r + half * 8) * G4_ + n] = (__bf16)acc[r];
    }

    // prefetch next step's xg row segments (hidden across barrier + elementwise)
    if (t + 1 < S_) {
      const float* nxt = xg + xgrow + G4_;
      __builtin_prefetch(nxt + ec, 0, 0);
      __builtin_prefetch(nxt + 256 + ec, 0, 0);
      __builtin_prefetch(nxt + 512 + ec, 0, 0);
      __builtin_prefetch(nxt + 768 + ec, 0, 0);
    }
    __syncthreads();   // gates visible; all h_lds reads done

    // ---- LSTM cell (gate order i, f, g, o) ----
    const size_t hrow = ((size_t)(b0 + er) * S_ + t) * H_;
#pragma unroll
    for (int j = 0; j < 16; ++j) {
      const int hc = ec + j;
      float ig = (float)g_lds[er * G4_ + hc]       + xv[0][j >> 2][j & 3];
      float fg = (float)g_lds[er * G4_ + 256 + hc] + xv[1][j >> 2][j & 3];
      float gg = (float)g_lds[er * G4_ + 512 + hc] + xv[2][j >> 2][j & 3];
      float og = (float)g_lds[er * G4_ + 768 + hc] + xv[3][j >> 2][j & 3];
      ig = sigf(ig); fg = sigf(fg); og = sigf(og); gg = tanhf(gg);
      const float c_new = fg * c_st[j] + ig * gg;
      c_st[j] = c_new;
      const float h = og * tanhf(c_new);
      const __bf16 hb = (__bf16)h;
      h_lds[er * HPAD + hc] = hb;
      hs[hrow + hc] = hb;
    }
    __syncthreads();   // h_lds visible for next step; g_lds reads done
  }
}

// ---------------- out = hs @ fc_W^T + fc_b ----------------
__global__ void k_fc(const __bf16* __restrict__ hs,   // [B*S, H]
                     const __bf16* __restrict__ fcw,  // [O, H]
                     const float*  __restrict__ fcb,
                     float*        __restrict__ out)  // [B*S, O]
{
  const int lane = threadIdx.x & 31;
  const int wv   = threadIdx.x >> 5;
  const int gid  = blockIdx.x * 8 + wv;
  const int M0   = (gid >> 3) * 16;   // 4096 M tiles
  const int N0   = (gid & 7) * 16;    // 8 N tiles
  const int half = lane >> 4;
  const int l15  = lane & 15;
  const int kaoff = half * 8;
  const int kboff = kaoff + 16;

  const __bf16* arow = hs  + (size_t)(M0 + l15) * H_;
  const __bf16* brow = fcw + (size_t)(N0 + l15) * H_;

  v8f acc = {};
#pragma unroll
  for (int kc = 0; kc < 8; ++kc) {
    v16bf a = load_frag(arow + kc * 32 + kaoff, arow + kc * 32 + kboff);
    v16bf b = load_frag(brow + kc * 32 + kaoff, brow + kc * 32 + kboff);
    acc = __builtin_amdgcn_wmma_f32_16x16x32_bf16(false, a, false, b, (short)0, acc, false, false);
  }
  const int n = N0 + l15;
  const float bias = fcb[n];
#pragma unroll
  for (int r = 0; r < 8; ++r)
    out[(size_t)(M0 + r + half * 8) * O_ + n] = acc[r] + bias;
}

extern "C" void kernel_launch(void* const* d_in, const int* in_sizes, int n_in,
                              void* d_out, int out_size, void* d_ws, size_t ws_size,
                              hipStream_t stream) {
  const float* x   = (const float*)d_in[0];
  const float* Wih = (const float*)d_in[1];
  const float* Whh = (const float*)d_in[2];
  const float* bih = (const float*)d_in[3];
  const float* bhh = (const float*)d_in[4];
  const float* fcW = (const float*)d_in[5];
  const float* fcb = (const float*)d_in[6];
  float* out = (float*)d_out;

  char* ws = (char*)d_ws;
  size_t off = 0;
  auto carve = [&](size_t bytes) -> char* {
    char* p = ws + off;
    off += (bytes + 255) & ~(size_t)255;
    return p;
  };
  __bf16* xbf  = (__bf16*)carve((size_t)B_ * S_ * I_ * 2);
  __bf16* wihb = (__bf16*)carve((size_t)G4_ * I_ * 2);
  __bf16* whhb = (__bf16*)carve((size_t)G4_ * H_ * 2);
  __bf16* fcwb = (__bf16*)carve((size_t)O_ * H_ * 2);
  float*  xg   = (float*)carve((size_t)B_ * S_ * G4_ * 4);
  __bf16* hsb  = (__bf16*)carve((size_t)B_ * S_ * H_ * 2);
  (void)ws_size; (void)in_sizes; (void)n_in; (void)out_size;

  k_cvt_bf16<<<(B_ * S_ * I_ + 255) / 256, 256, 0, stream>>>(x, xbf, B_ * S_ * I_);
  k_cvt_bf16<<<(G4_ * I_ + 255) / 256, 256, 0, stream>>>(Wih, wihb, G4_ * I_);
  k_cvt_bf16<<<(G4_ * H_ + 255) / 256, 256, 0, stream>>>(Whh, whhb, G4_ * H_);
  k_cvt_bf16<<<(O_ * H_ + 255) / 256, 256, 0, stream>>>(fcW, fcwb, O_ * H_);

  // xg GEMM: 4096x64 tiles, 8 waves/block
  k_xg_gemm<<<(4096 * 64) / 8, 256, 0, stream>>>(xbf, wihb, bih, bhh, xg);
  // recurrent scan: one workgroup per 16-row batch tile
  k_lstm_scan<<<B_ / 16, 256, 0, stream>>>(xg, whhb, hsb);
  // output FC: 4096x8 tiles
  k_fc<<<(4096 * 8) / 8, 256, 0, stream>>>(hsb, fcwb, fcb, out);
}
